// MyLRNNCell_65369402245632
// MI455X (gfx1250) — matrix operand order repaired and checked by reading, not compiled, not run on hardware
//
#include <hip/hip_runtime.h>
#include <hip/hip_bf16.h>
#include <math.h>

#define UNITS   256
#define TSTEPS  2048
#define BATCH   64
#define LROW    264           // 256 + 8 bf16 pad -> 528B row stride, conflict-free b128
#define EPS     0.01f
#define GAMMA   0.01f

typedef __attribute__((ext_vector_type(16))) __bf16 v16bf;
typedef __attribute__((ext_vector_type(8)))  __bf16 v8bf;
typedef __attribute__((ext_vector_type(8)))  float  v8f;

// ---- WMMA fragment loaders (LDS, bf16, padded-row layout) ----

// A-matrix 16x32 bf16: lanes 0-15 -> row M=lane, K = k0+{0..7, 16..23}
//                      lanes 16-31 -> row M=lane-16, K = k0+{8..15, 24..31}
static __device__ __forceinline__ v16bf load_fragA(const __bf16* base, int k0) {
  const int lane = threadIdx.x & 31;
  const int m  = lane & 15;
  const int kh = (lane >> 4) << 3;                 // 0 or 8
  const __bf16* p = base + m * LROW + k0 + kh;
  v8bf lo = *(const v8bf*)(p);                     // K = k0+kh .. +7   (ds_load_b128)
  v8bf hi = *(const v8bf*)(p + 16);                // K = k0+kh+16 .. +23
  return __builtin_shufflevector(lo, hi, 0,1,2,3,4,5,6,7,8,9,10,11,12,13,14,15);
}

// B-matrix 32x16 bf16 (stored transposed: baseT[n][k]):
// lanes 0-15 -> col N=lane, K = k0+{0..7, 16..23} ; lanes 16-31 -> col N=lane-16, K = k0+{8..15, 24..31}
static __device__ __forceinline__ v16bf load_fragB(const __bf16* baseT, int n0, int k0) {
  const int lane = threadIdx.x & 31;
  const int n  = lane & 15;
  const int kh = (lane >> 4) << 3;                 // 0 or 8
  const __bf16* p = baseT + (n0 + n) * LROW + k0 + kh;
  v8bf lo = *(const v8bf*)(p);
  v8bf hi = *(const v8bf*)(p + 16);
  return __builtin_shufflevector(lo, hi, 0,1,2,3,4,5,6,7,8,9,10,11,12,13,14,15);
}

static __device__ __forceinline__ v8f wmma_bf16(v16bf a, v16bf b, v8f c) {
  return __builtin_amdgcn_wmma_f32_16x16x32_bf16(false, a, false, b, (short)0, c,
                                                 false, false);
}

// =====================================================================
// Kernel 1: xu = x @ U + b  -> written into d_out (same shape as output)
// 1024 WGs x 256 threads; each WG does a 128x256 slab. Bandwidth-bound
// (268 MB total traffic ~ 11.5us at 23.3 TB/s); bf16 WMMA is overkill
// but free here.
// LDS: UT bf16 [256][LROW] (132KB) + Xb bf16 [128][LROW] (66KB) = 198KB
// =====================================================================
#define XU_LDS_BYTES ((256 * LROW + 128 * LROW) * 2)

__global__ void __launch_bounds__(256) xu_proj_kernel(
    const float* __restrict__ x, const float* __restrict__ U,
    const float* __restrict__ bvec, float* __restrict__ out) {
  extern __shared__ __align__(16) char lds_raw[];
  __bf16* UT = (__bf16*)lds_raw;          // UT[n][k] = U[k][n]
  __bf16* Xb = UT + 256 * LROW;           // Xb[i][k] = x[row0+i][k]

  const int tid   = threadIdx.x;
  const size_t row0 = (size_t)blockIdx.x * 128;

  {
    const int n = tid;                    // coalesced across threads per k
    for (int k = 0; k < 256; ++k)
      UT[n * LROW + k] = (__bf16)U[k * 256 + n];
  }
  for (int e = tid; e < 128 * 256; e += 256) {
    const int i = e >> 8, k = e & 255;
    Xb[i * LROW + k] = (__bf16)x[row0 * 256 + (size_t)e];
  }
  __syncthreads();

  const int wave  = tid >> 5;             // 0..7: M-tile
  const int lane  = tid & 31;
  const int lcol  = lane & 15;
  const int mbase = (lane >> 4) << 3;
  const __bf16* Arows = Xb + wave * 16 * LROW;

  v16bf af[8];
#pragma unroll
  for (int kc = 0; kc < 8; ++kc) af[kc] = load_fragA(Arows, kc * 32);

  for (int nt = 0; nt < 16; ++nt) {
    v8f acc = {};
#pragma unroll
    for (int kc = 0; kc < 8; ++kc)
      acc = wmma_bf16(af[kc], load_fragB(UT, nt * 16, kc * 32), acc);
    const float bias = bvec[nt * 16 + lcol];
#pragma unroll
    for (int r = 0; r < 8; ++r) {
      const size_t grow = row0 + (size_t)(wave * 16 + mbase + r);
      out[grow * 256 + nt * 16 + lcol] = acc[r] + bias;
    }
  }
}

// =====================================================================
// Kernel 2: sequential scan. 4 WGs x 256 threads (8 waves); WG g owns
// batch rows 16g..15+16g. All loop-invariant B-fragments (A^T and W^T
// tiles for this wave's two N-tiles) are hoisted into ~256 VGPRs for
// the entire 2048-step loop; h lives in 16 persistent VGPRs per lane
// (f32), with only a bf16 cross-lane copy in LDS for A-fragments.
// Per step per wave: 16 ds_load_b128 + 32 back-to-back WMMAs.
// LDS: AT (132KB) + WT (132KB) + Hb (8.25KB) = 272.25KB of the 320KB/WG.
// =====================================================================
#define SCAN_LDS_BYTES (2 * 256 * LROW * 2 + 16 * LROW * 2)

__global__ void __launch_bounds__(256, 2) scan_kernel(
    const float* __restrict__ Bm, const float* __restrict__ Cm,
    float* __restrict__ out) {
  extern __shared__ __align__(16) char lds_raw[];
  __bf16* AT = (__bf16*)lds_raw;                  // [256][LROW], AT[n][k]=A[k][n]
  __bf16* WT = AT + 256 * LROW;                   // [256][LROW]
  __bf16* Hb = WT + 256 * LROW;                   // [16][LROW] bf16 h for A-frags

  const int tid = threadIdx.x;
  const int b0  = blockIdx.x * 16;

  // Prologue: A = B - 0.6*B^T - g*I ; W = C - 0.6*C^T - g*I (beta=0.8), transposed.
  {
    const int n = tid;
    for (int k = 0; k < 256; ++k) {
      const float diag = (k == n) ? GAMMA : 0.0f;
      AT[n * LROW + k] = (__bf16)(Bm[k * 256 + n] - 0.6f * Bm[n * 256 + k] - diag);
      WT[n * LROW + k] = (__bf16)(Cm[k * 256 + n] - 0.6f * Cm[n * 256 + k] - diag);
    }
  }
  for (int i = tid; i < 16 * LROW; i += 256) Hb[i] = (__bf16)0.0f;
  __syncthreads();

  const int wave  = tid >> 5;
  const int lane  = tid & 31;
  const int lcol  = lane & 15;
  const int mbase = (lane >> 4) << 3;
  const int n1 = wave * 16;                        // this wave's two N-tiles
  const int n2 = (wave + 8) * 16;

  // Hoist ALL loop-invariant B-fragments into registers (4 streams x 8 kc).
  v16bf bA1[8], bW1[8], bA2[8], bW2[8];
#pragma unroll
  for (int kc = 0; kc < 8; ++kc) {
    bA1[kc] = load_fragB(AT, n1, kc * 32);
    bW1[kc] = load_fragB(WT, n1, kc * 32);
    bA2[kc] = load_fragB(AT, n2, kc * 32);
    bW2[kc] = load_fragB(WT, n2, kc * 32);
  }

  // Persistent f32 h state: each lane owns exactly the C/D elements it
  // updates ([m=mbase+r][n1|n2 + lcol]), so no f32 LDS copy is needed.
  float h1[8] = {0.f, 0.f, 0.f, 0.f, 0.f, 0.f, 0.f, 0.f};
  float h2[8] = {0.f, 0.f, 0.f, 0.f, 0.f, 0.f, 0.f, 0.f};

  for (int t = 0; t < TSTEPS; ++t) {
    const size_t toff = (size_t)t * 256;

    // Issue xu global loads early so they overlap the WMMA work (xu is
    // L2-resident: 134MB output < 192MB L2, freshly written by kernel 1).
    float xu1[8], xu2[8];
#pragma unroll
    for (int r = 0; r < 8; ++r) {
      const int m = mbase + r;
      const size_t g = ((size_t)(b0 + m) * TSTEPS) * 256 + toff;
      xu1[r] = out[g + n1 + lcol];
      xu2[r] = out[g + n2 + lcol];
    }

    // A-fragments of h: the only per-step LDS loads (16 ds_load_b128).
    v16bf af[8];
#pragma unroll
    for (int kc = 0; kc < 8; ++kc) af[kc] = load_fragA(Hb, kc * 32);

    // 32 WMMAs, 4 interleaved accumulation chains, B operands in regs.
    v8f aA1 = {}, aW1 = {}, aA2 = {}, aW2 = {};
#pragma unroll
    for (int kc = 0; kc < 8; ++kc) {
      aA1 = wmma_bf16(af[kc], bA1[kc], aA1);
      aW1 = wmma_bf16(af[kc], bW1[kc], aW1);
      aA2 = wmma_bf16(af[kc], bA2[kc], aA2);
      aW2 = wmma_bf16(af[kc], bW2[kc], aW2);
    }

    __syncthreads();  // all waves done reading Hb before overwrite

#pragma unroll
    for (int r = 0; r < 8; ++r) {
      const int m = mbase + r;
      const size_t g = ((size_t)(b0 + m) * TSTEPS) * 256 + toff;
      const float hn1 = h1[r] + EPS * aA1[r] + EPS * tanhf(aW1[r] + xu1[r]);
      const float hn2 = h2[r] + EPS * aA2[r] + EPS * tanhf(aW2[r] + xu2[r]);
      h1[r] = hn1;
      h2[r] = hn2;
      out[g + n1 + lcol] = hn1;                   // h_t overwrites xu_t in place
      out[g + n2 + lcol] = hn2;
      Hb[m * LROW + n1 + lcol] = (__bf16)hn1;
      Hb[m * LROW + n2 + lcol] = (__bf16)hn2;
    }
    __syncthreads();  // new Hb visible before next step's A-fragment loads
  }
}

// =====================================================================
extern "C" void kernel_launch(void* const* d_in, const int* in_sizes, int n_in,
                              void* d_out, int out_size, void* d_ws, size_t ws_size,
                              hipStream_t stream) {
  (void)in_sizes; (void)n_in; (void)out_size; (void)d_ws; (void)ws_size;
  // setup_inputs order: x, C, B, U, b
  const float* x  = (const float*)d_in[0];
  const float* Cm = (const float*)d_in[1];
  const float* Bm = (const float*)d_in[2];
  const float* U  = (const float*)d_in[3];
  const float* bv = (const float*)d_in[4];
  float* out = (float*)d_out;

  hipFuncSetAttribute((const void*)xu_proj_kernel,
                      hipFuncAttributeMaxDynamicSharedMemorySize, XU_LDS_BYTES);
  hipFuncSetAttribute((const void*)scan_kernel,
                      hipFuncAttributeMaxDynamicSharedMemorySize, SCAN_LDS_BYTES);

  // Phase 1: d_out = x @ U + b  (131072 rows / 128 per WG)
  xu_proj_kernel<<<(BATCH * TSTEPS) / 128, 256, XU_LDS_BYTES, stream>>>(x, U, bv, out);
  // Phase 2: in-place sequential scan over T, one WG per 16 batch rows.
  scan_kernel<<<BATCH / 16, 256, SCAN_LDS_BYTES, stream>>>(Bm, Cm, out);
}
